// TimbreTransformer_86311662780849
// MI455X (gfx1250) — matrix operand order repaired and checked by reading, not compiled, not run on hardware
//
#include <hip/hip_runtime.h>
#include <math.h>

#define D_MODEL 512
#define SEQ     2048
#define BATCH   4
#define NH      8
#define HD      64
#define DFF     2048
#define TOK     (BATCH*SEQ)   // 8192 rows

typedef __bf16 bf16;
typedef __bf16 v16bf __attribute__((ext_vector_type(16)));
typedef float  v8f   __attribute__((ext_vector_type(8)));
typedef unsigned int v4u  __attribute__((ext_vector_type(4)));
typedef unsigned int u32x4 __attribute__((ext_vector_type(4)));
typedef int          i32x4 __attribute__((ext_vector_type(4)));
typedef int          i32x8 __attribute__((ext_vector_type(8)));

union Frag { v16bf v; v4u q[2]; };

static __device__ __forceinline__ bf16 f2bf(float f) {
  union { float f; unsigned u; } a; a.f = f;
  unsigned r = (a.u + 0x7FFFu + ((a.u >> 16) & 1u)) >> 16;
  union { unsigned short s; bf16 b; } o; o.s = (unsigned short)r;
  return o.b;
}

static __device__ __forceinline__ v8f wmma_bf16(const Frag& A, const Frag& B, v8f C) {
  return __builtin_amdgcn_wmma_f32_16x16x32_bf16(false, A.v, false, B.v,
                                                 (short)0, C, false, false);
}

// ---------------------------------------------------------------------------
// Tensor Data Mover: 2D tile load global->LDS with LDS row padding.
//   tile_w elements of data_size=2B per row, tile_h rows,
//   global row stride = row_stride elements,
//   pad_interval: pad after 2^(v+1) DWORDs, pad_amount: (v+1) DWORDs.
// ---------------------------------------------------------------------------
static __device__ __forceinline__ void tdm_load_2d(unsigned lds_addr,
                                                   const void* gaddr,
                                                   unsigned tile_w,
                                                   unsigned tile_h,
                                                   unsigned row_stride,
                                                   unsigned pad_interval,
                                                   unsigned pad_amount) {
  unsigned long long ga = (unsigned long long)(uintptr_t)gaddr;
  u32x4 g0;
  g0[0] = 1u;                                   // count=1 (valid), user mode
  g0[1] = lds_addr;                             // LDS byte address
  g0[2] = (unsigned)ga;                         // global addr lo
  g0[3] = (unsigned)((ga >> 32) & 0x1FFFFFFu) | (2u << 30);  // addr hi | type=2
  const unsigned td0 = 1u << 20;                // huge tensor dims: no OOB clip
  const unsigned td1 = 1u << 20;
  i32x8 g1;
  g1[0] = (int)((1u << 16)                      // data_size = 2 bytes
              | (1u << 20)                      // pad_enable
              | (pad_interval << 22)
              | (pad_amount << 25));
  g1[1] = (int)((td0 & 0xFFFFu) << 16);         // abar_addr=0 | tensor_dim0 lo
  g1[2] = (int)((td0 >> 16) | ((td1 & 0xFFFFu) << 16));
  g1[3] = (int)((td1 >> 16) | (tile_w << 16));  // tensor_dim1 hi | tile_dim0
  g1[4] = (int)(tile_h & 0xFFFFu);              // tile_dim1 | tile_dim2=0
  g1[5] = (int)row_stride;                      // tensor_dim0_stride lo32
  g1[6] = 0;                                    // stride hi | dim1_stride lo
  g1[7] = 0;
  i32x4 z4 = {0, 0, 0, 0};
#if __has_include(<hip/amd_detail/amd_gfx1250_TDM.h>)
  i32x8 z8 = {0, 0, 0, 0, 0, 0, 0, 0};
  __builtin_amdgcn_tensor_load_to_lds(g0, g1, z4, z4, z8, 0);
#else
  __builtin_amdgcn_tensor_load_to_lds(g0, g1, z4, z4, 0);
#endif
}

static __device__ __forceinline__ unsigned lds_off(const void* p) {
  return (unsigned)(uintptr_t)p;   // generic LDS pointer: low 32 bits = LDS addr
}

// ---------------------------------------------------------------------------
// Weight convert + transpose: Wt[n*K + k] = bf16(W[k*N + n])
// ---------------------------------------------------------------------------
__global__ void k_wtrans(const float* __restrict__ W, bf16* __restrict__ Wt,
                         int K, int N) {
  int i = blockIdx.x * 256 + threadIdx.x;     // over N*K
  if (i >= N * K) return;
  int k = i % K, n = i / K;
  Wt[i] = f2bf(W[(size_t)k * N + n]);
}

// ---------------------------------------------------------------------------
// h = x + positional encoding (sin/cos interleaved)
// ---------------------------------------------------------------------------
__global__ void k_posenc(const float* __restrict__ x, float* __restrict__ h) {
  int i = blockIdx.x * 256 + threadIdx.x;     // over TOK*512
  int d = i & (D_MODEL - 1);
  int s = (i >> 9) & (SEQ - 1);
  int d2 = d & ~1;
  float ang = (float)s * __expf((float)d2 * (-9.210340371976184f / (float)D_MODEL));
  float pe = (d & 1) ? cosf(ang) : sinf(ang);
  h[i] = x[i] + pe;
}

// ---------------------------------------------------------------------------
// LayerNorm over D=512; one block per row; bf16 and/or f32 output
// ---------------------------------------------------------------------------
__global__ __launch_bounds__(256) void k_ln(const float* __restrict__ in,
                                            const float* __restrict__ g,
                                            const float* __restrict__ bt,
                                            bf16* __restrict__ outB,
                                            float* __restrict__ outF) {
  int row = blockIdx.x, t = threadIdx.x;
  const float* r = in + (size_t)row * D_MODEL;
  float v0 = r[t], v1 = r[t + 256];
  float s = v0 + v1, ss = v0 * v0 + v1 * v1;
  #pragma unroll
  for (int off = 16; off; off >>= 1) {
    s  += __shfl_xor(s,  off, 32);
    ss += __shfl_xor(ss, off, 32);
  }
  __shared__ float rs[8], rss[8];
  int w = t >> 5;
  if ((t & 31) == 0) { rs[w] = s; rss[w] = ss; }
  __syncthreads();
  float tot = 0.f, tot2 = 0.f;
  #pragma unroll
  for (int i = 0; i < 8; ++i) { tot += rs[i]; tot2 += rss[i]; }
  float mu  = tot * (1.0f / D_MODEL);
  float var = tot2 * (1.0f / D_MODEL) - mu * mu;
  float rr  = rsqrtf(var + 1e-5f);
  float y0 = (v0 - mu) * rr * g[t] + bt[t];
  float y1 = (v1 - mu) * rr * g[t + 256] + bt[t + 256];
  size_t base = (size_t)row * D_MODEL;
  if (outB) { outB[base + t] = f2bf(y0); outB[base + t + 256] = f2bf(y1); }
  if (outF) { outF[base + t] = y0;       outF[base + t + 256] = y1; }
}

// ---------------------------------------------------------------------------
// vT[(bh*64+d)*SEQ + s] = v[(b*SEQ+s)*512 + h*64 + d]
// ---------------------------------------------------------------------------
__global__ void k_vtrans(const bf16* __restrict__ v, bf16* __restrict__ vT) {
  int i = blockIdx.x * 256 + threadIdx.x;   // over BATCH*NH*HD*SEQ = 4M
  int s = i & (SEQ - 1);
  int r = i >> 11;                          // bh*64 + d
  int d = r & 63, bh = r >> 6;
  int b = bh >> 3, h = bh & 7;
  vT[i] = v[((size_t)(b * SEQ + s)) * D_MODEL + h * HD + d];
}

// ---------------------------------------------------------------------------
// GEMM: C[M,N] = A[M,K](bf16) @ Bt[N,K](bf16)^T + bias, epilogue fused.
// Block: 256 thr = 8 waves; tile 128x64. TDM double-buffered LDS staging:
// wave 0 issues tensor_load_to_lds for tile t+1, waits TENSORcnt<=2 for tile t.
// ---------------------------------------------------------------------------
__global__ __launch_bounds__(256) void k_gemm(const bf16* __restrict__ A,
                                              const bf16* __restrict__ Bt,
                                              const float* __restrict__ bias,
                                              const float* __restrict__ residual,
                                              float* __restrict__ outF,
                                              bf16* __restrict__ outB,
                                              int M, int N, int K, int gelu) {
  __shared__ __align__(16) bf16 lA[2][128][40];  // 32 data + 8 pad per row
  __shared__ __align__(16) bf16 lB[2][64][40];
  const int tid = threadIdx.x;
  const int w = tid >> 5, lane = tid & 31, hf = lane >> 4, l16 = lane & 15;
  const int m0 = blockIdx.y * 128, n0 = blockIdx.x * 64;
  const int nk = K >> 5;
  v8f c[4] = {};

  if (w == 0) {   // prologue: fetch tile 0
    tdm_load_2d(lds_off(&lA[0][0][0]), A + (size_t)m0 * K, 32, 128, K, 3, 3);
    tdm_load_2d(lds_off(&lB[0][0][0]), Bt + (size_t)n0 * K, 32, 64, K, 3, 3);
  }

  for (int kt = 0; kt < nk; ++kt) {
    const int buf = kt & 1;
    if (w == 0) {
      if (kt + 1 < nk) {   // prefetch next tile into other buffer
        int k1 = (kt + 1) << 5;
        tdm_load_2d(lds_off(&lA[buf ^ 1][0][0]), A + (size_t)m0 * K + k1,
                    32, 128, K, 3, 3);
        tdm_load_2d(lds_off(&lB[buf ^ 1][0][0]), Bt + (size_t)n0 * K + k1,
                    32, 64, K, 3, 3);
        __builtin_amdgcn_s_wait_tensorcnt((short)2);  // tile kt complete
      } else {
        __builtin_amdgcn_s_wait_tensorcnt((short)0);
      }
    }
    __syncthreads();

    Frag a;
    const bf16* pa = &lA[buf][w * 16 + l16][0];
    a.q[0] = *(const v4u*)(pa + 8 * hf);
    a.q[1] = *(const v4u*)(pa + 16 + 8 * hf);
    #pragma unroll
    for (int nt = 0; nt < 4; ++nt) {
      Frag b;
      const bf16* pb = &lB[buf][nt * 16 + l16][0];
      b.q[0] = *(const v4u*)(pb + 8 * hf);
      b.q[1] = *(const v4u*)(pb + 16 + 8 * hf);
      c[nt] = wmma_bf16(a, b, c[nt]);
    }
    __syncthreads();   // reads done before TDM overwrites this buffer
  }

  // epilogue: C layout -> lane = col (l16), VGPR j -> row j + 8*hf
  #pragma unroll
  for (int nt = 0; nt < 4; ++nt) {
    int gn = n0 + nt * 16 + l16;
    float bi = bias[gn];
    #pragma unroll
    for (int j = 0; j < 8; ++j) {
      int gm = m0 + w * 16 + hf * 8 + j;
      float v = c[nt][j] + bi;
      if (gelu) v = 0.5f * v * (1.0f + erff(v * 0.7071067811865475f));
      size_t idx = (size_t)gm * N + gn;
      if (residual) v += residual[idx];
      if (outF) outF[idx] = v;
      if (outB) outB[idx] = f2bf(v);
    }
  }
}

// ---------------------------------------------------------------------------
// Flash attention: block = 128 query rows x one (b,h); streaming softmax.
// K-block and V^T-block staged via TDM, double buffered.
// ---------------------------------------------------------------------------
__global__ __launch_bounds__(256) void k_attn(const bf16* __restrict__ Q,
                                              const bf16* __restrict__ Km,
                                              const bf16* __restrict__ VT,
                                              bf16* __restrict__ ctx) {
  __shared__ __align__(16) bf16 lK[2][64][72];   // 64 data + 8 pad per row
  __shared__ __align__(16) bf16 lV[2][64][72];
  __shared__ __align__(16) bf16 lP[8][16][72];
  const int tid = threadIdx.x;
  const int w = tid >> 5, lane = tid & 31, hf = lane >> 4, l16 = lane & 15;
  const int s0 = blockIdx.x * 128;
  const int bh = blockIdx.y, b = bh >> 3, h = bh & 7;

  const bf16* qrow  = Q  + ((size_t)(b * SEQ + s0 + w * 16 + l16)) * D_MODEL + h * HD;
  const bf16* kbase = Km + ((size_t)b * SEQ) * D_MODEL + h * HD;
  const bf16* vtb   = VT + (size_t)bh * HD * SEQ;

  v8f o[4] = {};
  float mrow[8], lrow[8];
  #pragma unroll
  for (int j = 0; j < 8; ++j) { mrow[j] = -1e30f; lrow[j] = 0.f; }

  // Q fragments are loop-invariant (d = 64 -> two 32-wide K-steps)
  Frag qa[2];
  #pragma unroll
  for (int ks = 0; ks < 2; ++ks) {
    qa[ks].q[0] = *(const v4u*)(qrow + ks * 32 + 8 * hf);
    qa[ks].q[1] = *(const v4u*)(qrow + ks * 32 + 16 + 8 * hf);
  }

  const int nkb = SEQ / 64;   // 32 key blocks
  if (w == 0) {   // prologue: key block 0
    tdm_load_2d(lds_off(&lK[0][0][0]), kbase, 64, 64, D_MODEL, 4, 3);
    tdm_load_2d(lds_off(&lV[0][0][0]), vtb,   64, 64, SEQ,     4, 3);
  }

  for (int kt = 0; kt < nkb; ++kt) {
    const int buf = kt & 1;
    if (w == 0) {
      if (kt + 1 < nkb) {
        int kb1 = (kt + 1) * 64;
        tdm_load_2d(lds_off(&lK[buf ^ 1][0][0]),
                    kbase + (size_t)kb1 * D_MODEL, 64, 64, D_MODEL, 4, 3);
        tdm_load_2d(lds_off(&lV[buf ^ 1][0][0]),
                    vtb + kb1, 64, 64, SEQ, 4, 3);
        __builtin_amdgcn_s_wait_tensorcnt((short)2);
      } else {
        __builtin_amdgcn_s_wait_tensorcnt((short)0);
      }
    }
    __syncthreads();

    // scores: S = Q @ K^T  (16x64 per wave)
    v8f s[4] = {};
    #pragma unroll
    for (int ks = 0; ks < 2; ++ks) {
      #pragma unroll
      for (int nt = 0; nt < 4; ++nt) {
        Frag bq;
        const bf16* kr = &lK[buf][nt * 16 + l16][ks * 32];
        bq.q[0] = *(const v4u*)(kr + 8 * hf);
        bq.q[1] = *(const v4u*)(kr + 16 + 8 * hf);
        s[nt] = wmma_bf16(qa[ks], bq, s[nt]);
      }
    }
    #pragma unroll
    for (int nt = 0; nt < 4; ++nt)
      #pragma unroll
      for (int j = 0; j < 8; ++j) s[nt][j] *= 0.125f;   // 1/sqrt(64)

    // streaming softmax update (rows live per (hf, j); reduce over 16 lanes)
    #pragma unroll
    for (int j = 0; j < 8; ++j) {
      float mx = fmaxf(fmaxf(s[0][j], s[1][j]), fmaxf(s[2][j], s[3][j]));
      #pragma unroll
      for (int off = 1; off < 16; off <<= 1) mx = fmaxf(mx, __shfl_xor(mx, off, 16));
      float mn = fmaxf(mrow[j], mx);
      float corr = __expf(mrow[j] - mn);
      mrow[j] = mn;
      #pragma unroll
      for (int nt = 0; nt < 4; ++nt) o[nt][j] *= corr;
      float rsum = 0.f;
      #pragma unroll
      for (int nt = 0; nt < 4; ++nt) {
        float p = __expf(s[nt][j] - mn);
        s[nt][j] = p;
        rsum += p;
      }
      #pragma unroll
      for (int off = 1; off < 16; off <<= 1) rsum += __shfl_xor(rsum, off, 16);
      lrow[j] = lrow[j] * corr + rsum;
    }

    // P: C-layout -> A-layout via wave-private LDS region
    #pragma unroll
    for (int nt = 0; nt < 4; ++nt)
      #pragma unroll
      for (int j = 0; j < 8; ++j)
        lP[w][hf * 8 + j][nt * 16 + l16] = f2bf(s[nt][j]);

    // O += P @ V  (contraction over 64 keys)
    #pragma unroll
    for (int ks = 0; ks < 2; ++ks) {
      Frag pa;
      const bf16* pp = &lP[w][l16][0];
      pa.q[0] = *(const v4u*)(pp + ks * 32 + 8 * hf);
      pa.q[1] = *(const v4u*)(pp + ks * 32 + 16 + 8 * hf);
      #pragma unroll
      for (int nt = 0; nt < 4; ++nt) {
        Frag vb;
        const bf16* vr = &lV[buf][nt * 16 + l16][ks * 32];
        vb.q[0] = *(const v4u*)(vr + 8 * hf);
        vb.q[1] = *(const v4u*)(vr + 16 + 8 * hf);
        o[nt] = wmma_bf16(pa, vb, o[nt]);
      }
    }
    __syncthreads();   // all reads of this buffer done before TDM reuse
  }

  // normalize and write ctx (bf16, [B,S,512] head slice)
  #pragma unroll
  for (int j = 0; j < 8; ++j) {
    float inv = 1.0f / lrow[j];
    size_t gm = (size_t)(b * SEQ + s0 + w * 16 + hf * 8 + j);
    #pragma unroll
    for (int nt = 0; nt < 4; ++nt)
      ctx[gm * D_MODEL + h * HD + nt * 16 + l16] = f2bf(o[nt][j] * inv);
  }
}

// ---------------------------------------------------------------------------
extern "C" void kernel_launch(void* const* d_in, const int* in_sizes, int n_in,
                              void* d_out, int out_size, void* d_ws, size_t ws_size,
                              hipStream_t stream) {
  (void)in_sizes; (void)n_in; (void)out_size; (void)ws_size;
  const float* x   = (const float*)d_in[0];
  const float* wq  = (const float*)d_in[1];
  const float* bq  = (const float*)d_in[2];
  const float* wk  = (const float*)d_in[3];
  const float* bk  = (const float*)d_in[4];
  const float* wv  = (const float*)d_in[5];
  const float* bv  = (const float*)d_in[6];
  const float* wo  = (const float*)d_in[7];
  const float* bo  = (const float*)d_in[8];
  const float* w1  = (const float*)d_in[9];
  const float* b1  = (const float*)d_in[10];
  const float* w2  = (const float*)d_in[11];
  const float* b2  = (const float*)d_in[12];
  const float* g1  = (const float*)d_in[13];
  const float* be1 = (const float*)d_in[14];
  const float* g2  = (const float*)d_in[15];
  const float* be2 = (const float*)d_in[16];
  const float* gf  = (const float*)d_in[17];
  const float* bef = (const float*)d_in[18];

  char* p = (char*)d_ws;
  auto take = [&](size_t bytes) -> char* {
    char* r = p; p += (bytes + 255) & ~(size_t)255; return r;
  };
  float* h   = (float*)take((size_t)TOK * D_MODEL * 4);
  bf16* x1   = (bf16*)take((size_t)TOK * D_MODEL * 2);
  bf16* q    = (bf16*)take((size_t)TOK * D_MODEL * 2);
  bf16* km   = (bf16*)take((size_t)TOK * D_MODEL * 2);
  bf16* v    = (bf16*)take((size_t)TOK * D_MODEL * 2);
  bf16* vT   = (bf16*)take((size_t)TOK * D_MODEL * 2);
  bf16* ctx  = (bf16*)take((size_t)TOK * D_MODEL * 2);
  bf16* ff   = (bf16*)take((size_t)TOK * DFF * 2);
  bf16* wqT  = (bf16*)take((size_t)D_MODEL * D_MODEL * 2);
  bf16* wkT  = (bf16*)take((size_t)D_MODEL * D_MODEL * 2);
  bf16* wvT  = (bf16*)take((size_t)D_MODEL * D_MODEL * 2);
  bf16* woT  = (bf16*)take((size_t)D_MODEL * D_MODEL * 2);
  bf16* w1T  = (bf16*)take((size_t)D_MODEL * DFF * 2);
  bf16* w2T  = (bf16*)take((size_t)DFF * D_MODEL * 2);

  int nW = D_MODEL * D_MODEL;
  k_wtrans<<<(nW + 255) / 256, 256, 0, stream>>>(wq, wqT, D_MODEL, D_MODEL);
  k_wtrans<<<(nW + 255) / 256, 256, 0, stream>>>(wk, wkT, D_MODEL, D_MODEL);
  k_wtrans<<<(nW + 255) / 256, 256, 0, stream>>>(wv, wvT, D_MODEL, D_MODEL);
  k_wtrans<<<(nW + 255) / 256, 256, 0, stream>>>(wo, woT, D_MODEL, D_MODEL);
  int nW1 = D_MODEL * DFF;
  k_wtrans<<<(nW1 + 255) / 256, 256, 0, stream>>>(w1, w1T, D_MODEL, DFF);
  k_wtrans<<<(nW1 + 255) / 256, 256, 0, stream>>>(w2, w2T, DFF, D_MODEL);

  k_posenc<<<(TOK * D_MODEL) / 256, 256, 0, stream>>>(x, h);

  dim3 gD(D_MODEL / 64, TOK / 128);   // (8, 64)  N=512 GEMMs
  dim3 gF(DFF / 64, TOK / 128);       // (32, 64) N=2048 GEMM
  dim3 gA(SEQ / 128, BATCH * NH);     // (16, 32) attention

  for (int layer = 0; layer < 6; ++layer) {
    k_ln<<<TOK, 256, 0, stream>>>(h, g1, be1, x1, nullptr);
    k_gemm<<<gD, 256, 0, stream>>>(x1, wqT, bq, nullptr, nullptr, q,
                                   TOK, D_MODEL, D_MODEL, 0);
    k_gemm<<<gD, 256, 0, stream>>>(x1, wkT, bk, nullptr, nullptr, km,
                                   TOK, D_MODEL, D_MODEL, 0);
    k_gemm<<<gD, 256, 0, stream>>>(x1, wvT, bv, nullptr, nullptr, v,
                                   TOK, D_MODEL, D_MODEL, 0);
    k_vtrans<<<(BATCH * NH * HD * SEQ) / 256, 256, 0, stream>>>(v, vT);
    k_attn<<<gA, 256, 0, stream>>>(q, km, vT, ctx);
    k_gemm<<<gD, 256, 0, stream>>>(ctx, woT, bo, h, h, nullptr,
                                   TOK, D_MODEL, D_MODEL, 0);
    k_ln<<<TOK, 256, 0, stream>>>(h, g2, be2, x1, nullptr);
    k_gemm<<<gF, 256, 0, stream>>>(x1, w1T, b1, nullptr, nullptr, ff,
                                   TOK, DFF, D_MODEL, 1);
    k_gemm<<<gD, 256, 0, stream>>>(ff, w2T, b2, h, h, nullptr,
                                   TOK, D_MODEL, DFF, 0);
  }
  k_ln<<<TOK, 256, 0, stream>>>(h, gf, bef, nullptr, (float*)d_out);
}
